// base_model_69423851373102
// MI455X (gfx1250) — matrix operand order repaired and checked by reading, not compiled, not run on hardware
//
#include <hip/hip_runtime.h>
#include <hip/hip_bf16.h>

// ---------------------------------------------------------------------------
// Types / WMMA helpers (CDNA5 gfx1250, wave32)
// ---------------------------------------------------------------------------
typedef __bf16 bf16_t;
typedef __attribute__((ext_vector_type(16))) __bf16 v16bf;
typedef __attribute__((ext_vector_type(8)))  __bf16 v8bf;
typedef __attribute__((ext_vector_type(8)))  float  v8f;

#define NR 8192          // batch / node count
#define MT (NR / 16)     // 512 row tiles

__device__ __forceinline__ v8f vzero() {
  v8f z = {0.f,0.f,0.f,0.f,0.f,0.f,0.f,0.f};
  return z;
}

// Wave-uniform value in an SGPR (keeps EXEC manipulation out of WMMA paths).
__device__ __forceinline__ int uniform_wid() {
  int w = (int)((blockIdx.x * blockDim.x + threadIdx.x) >> 5);
  return __builtin_amdgcn_readfirstlane(w);
}

// A-fragment (16x32 bf16): lane l16 = row M, half h = lane>>4.
// elements 0..15 -> K = k0 + 16h + e (contiguous)
__device__ __forceinline__ v16bf load_frag_a(const bf16_t* p) {
  v8bf lo = *(const v8bf*)(p);
  v8bf hi = *(const v8bf*)(p + 8);
  v16bf r;
#pragma unroll
  for (int i = 0; i < 8; ++i) { r[i] = lo[i]; r[i + 8] = hi[i]; }
  return r;
}

// B-fragment (32x16 bf16): lane l16 = column N, half h = lane>>4.
// 16 contiguous K values starting at k0 + 16h  (one 32-byte load)
__device__ __forceinline__ v16bf load_frag_b(const bf16_t* p) {
  return *(const v16bf*)(p);
}

__device__ __forceinline__ v8f wmma_bf16(v16bf a, v16bf b, v8f c) {
  return __builtin_amdgcn_wmma_f32_16x16x32_bf16(false, a, false, b, (short)0, c,
                                                 false, false);
}

__device__ __forceinline__ float wave_sum(float v) {
#pragma unroll
  for (int off = 16; off > 0; off >>= 1) v += __shfl_xor(v, off, 32);
  return v;
}

// ---------------------------------------------------------------------------
// Elementwise / utility kernels
// ---------------------------------------------------------------------------
__global__ void k_fill(float* p, long long n, float v) {
  long long i = (long long)blockIdx.x * blockDim.x + threadIdx.x;
  if (i < n) p[i] = v;
}

// f32 [rows, srcLd] (first K cols) -> bf16 [rows, Kp] zero-padded
__global__ void k_cvt_pad(const float* __restrict__ src, int srcLd, int K,
                          bf16_t* __restrict__ dst, int Kp, long long total) {
  long long i = (long long)blockIdx.x * blockDim.x + threadIdx.x;
  if (i >= total) return;
  int       c = (int)(i % Kp);
  long long r = i / Kp;
  float v = (c < K) ? src[r * (long long)srcLd + c] : 0.f;
  dst[i] = (bf16_t)v;
}

// ---------------------------------------------------------------------------
// Staged WMMA GEMM (Fout % 64 == 0):
//   D[M,Fout] = A_bf16[M,Kp] @ W_bf16[Fout,Kp]^T (+ bias)
// Block = 8 waves = 8 consecutive 16-row M tiles sharing one 64-col N group.
// The 64x32 bf16 B panel is double-buffered in LDS via async global->LDS
// copies (ASYNCcnt), overlapping the copy of panel k+1 with WMMA on panel k.
// All control flow is scalar: EXEC is all-ones at every WMMA.
// ---------------------------------------------------------------------------
__global__ void __launch_bounds__(256)
k_gemm4s(const bf16_t* __restrict__ A, const bf16_t* __restrict__ W,
         const float* __restrict__ bias, float* __restrict__ D,
         int Kp, int ldD, int colOff) {
  __shared__ __align__(16) bf16_t panel[2][64 * 32];   // 2 x 4KB

  const int blk    = __builtin_amdgcn_readfirstlane((int)blockIdx.x);
  const int mtBase = (blk % (MT / 8)) * 8;
  const int ng     = blk / (MT / 8);
  const int tid    = threadIdx.x;
  const int wave   = __builtin_amdgcn_readfirstlane(tid >> 5);
  const int lane   = tid & 31, l16 = lane & 15, h = lane >> 4;
  const int mt     = mtBase + wave;

  // async-copy mapping: thread t copies 16B: panel row t/4, quarter t%4
  const int crow = tid >> 2;
  const int cq   = tid & 3;
  const bf16_t* gsrc = W + (size_t)(ng * 64 + crow) * Kp + cq * 8;
  const unsigned ldsRowOff = (unsigned)(crow * 64 + cq * 16);

  v8f acc[4];
#pragma unroll
  for (int t = 0; t < 4; ++t) acc[t] = vzero();

  const bf16_t* aPtr = A + (size_t)(mt * 16 + l16) * Kp + h * 16;

  {  // prologue: stage panel 0
    unsigned dst = (unsigned)(uintptr_t)(&panel[0][0]) + ldsRowOff;
    asm volatile("global_load_async_to_lds_b128 %0, %1, off"
                 :: "v"(dst), "v"(gsrc) : "memory");
  }

  int cur = 0;
  for (int k0 = 0; k0 < Kp; k0 += 32) {
    const int k1 = k0 + 32;
    if (k1 < Kp) {                                   // scalar branch
      unsigned dst = (unsigned)(uintptr_t)(&panel[cur ^ 1][0]) + ldsRowOff;
      const bf16_t* src = gsrc + k1;
      asm volatile("global_load_async_to_lds_b128 %0, %1, off"
                   :: "v"(dst), "v"(src) : "memory");
      asm volatile("s_wait_asynccnt 0x1" ::: "memory");   // own cur copy done
    } else {
      asm volatile("s_wait_asynccnt 0x0" ::: "memory");
    }
    __syncthreads();                                 // everyone's cur copy done

    v16bf af = load_frag_a(aPtr + k0);
    const bf16_t* pb = &panel[cur][0];
#pragma unroll
    for (int t = 0; t < 4; ++t) {
      v16bf bfr = load_frag_b(pb + (t * 16 + l16) * 32 + h * 16);
      acc[t] = wmma_bf16(af, bfr, acc[t]);
    }
    __syncthreads();                                 // cur free for overwrite
    cur ^= 1;
  }

#pragma unroll
  for (int t = 0; t < 4; ++t) {
    int col = (ng * 4 + t) * 16 + l16;
    float bv = bias ? bias[col] : 0.f;
#pragma unroll
    for (int e = 0; e < 8; ++e) {
      int row = mt * 16 + h * 8 + e;                 // D layout: m = e + 8h
      D[(size_t)row * ldD + colOff + col] = acc[t][e] + bv;
    }
  }
}

// Direct-load GEMM for the tiny Fout=16 case (NT compile-time, no guards).
template <int NT>
__global__ void k_gemm(const bf16_t* __restrict__ A, const bf16_t* __restrict__ W,
                       const float* __restrict__ bias, float* __restrict__ D,
                       int Kp, int ldD, int colOff, int totalWaves) {
  const int wid = uniform_wid();
  if (wid >= totalWaves) return;                    // scalar branch
  const int lane = threadIdx.x & 31;
  const int l16  = lane & 15;
  const int h    = lane >> 4;
  const int mt   = wid % MT;
  const int ng   = wid / MT;

  v8f acc[NT];
#pragma unroll
  for (int t = 0; t < NT; ++t) acc[t] = vzero();

  const bf16_t* aPtr = A + (size_t)(mt * 16 + l16) * Kp + h * 16;
  const bf16_t* bPtr = W + (size_t)(ng * NT * 16 + l16) * Kp + h * 16;
  for (int k0 = 0; k0 < Kp; k0 += 32) {
    v16bf af = load_frag_a(aPtr + k0);
#pragma unroll
    for (int t = 0; t < NT; ++t) {
      v16bf bfr = load_frag_b(bPtr + (size_t)t * 16 * Kp + k0);
      acc[t] = wmma_bf16(af, bfr, acc[t]);
    }
  }
#pragma unroll
  for (int t = 0; t < NT; ++t) {
    int col = (ng * NT + t) * 16 + l16;
    float bv = bias ? bias[col] : 0.f;
#pragma unroll
    for (int e = 0; e < 8; ++e) {
      int row = mt * 16 + h * 8 + e;
      D[(size_t)row * ldD + colOff + col] = acc[t][e] + bv;
    }
  }
}

// ---------------------------------------------------------------------------
// BatchNorm (training-mode batch stats, biased variance)
// ---------------------------------------------------------------------------
__global__ void k_bn_stats(const float* __restrict__ X, int F,
                           float* __restrict__ sum, float* __restrict__ sq) {
  int f  = threadIdx.x;                // blockDim.x == F
  int r0 = blockIdx.x * 64;
  float s = 0.f, q = 0.f;
  for (int r = 0; r < 64; ++r) {
    float v = X[(size_t)(r0 + r) * F + f];
    s += v; q += v * v;
  }
  atomicAdd(&sum[f], s);
  atomicAdd(&sq[f], q);
}

// aff[f] = g*rsqrt(var+eps), aff[256+f] = b - aff[f]*mean
__global__ void k_bn_fin(const float* sum, const float* sq, const float* g,
                         const float* b, int F, float* aff) {
  int f = threadIdx.x;
  if (f >= F) return;
  const float invM = 1.f / (float)NR;
  float m = sum[f] * invM;
  float v = sq[f] * invM - m * m;
  float a = g[f] * rsqrtf(v + 1e-5f);
  aff[f]       = a;
  aff[256 + f] = b[f] - a * m;
}

__global__ void k_bn_apply(const float* __restrict__ X, const float* __restrict__ aff,
                           int F, long long total, int relu,
                           float* __restrict__ outF, bf16_t* __restrict__ outB) {
  long long i = (long long)blockIdx.x * blockDim.x + threadIdx.x;
  if (i >= total) return;
  int f = (int)(i % F);
  float v = aff[f] * X[i] + aff[256 + f];
  if (relu) v = fmaxf(v, 0.f);
  if (outF) outF[i] = v;
  if (outB) outB[i] = (bf16_t)v;
}

// out = relu( bn2(T2) + res ), res = bn3(R) or identity R
__global__ void k_res_tail(const float* __restrict__ T2, const float* __restrict__ aff2,
                           const float* __restrict__ R, const float* __restrict__ aff3,
                           int F, long long total,
                           float* __restrict__ outF, bf16_t* __restrict__ outB) {
  long long i = (long long)blockIdx.x * blockDim.x + threadIdx.x;
  if (i >= total) return;
  int f = (int)(i % F);
  float v = aff2[f] * T2[i] + aff2[256 + f];
  float r = aff3 ? (aff3[f] * R[i] + aff3[256 + f]) : R[i];
  v = fmaxf(v + r, 0.f);
  outF[i] = v;
  outB[i] = (bf16_t)v;
}

// ---------------------------------------------------------------------------
// Graph stage
// ---------------------------------------------------------------------------
// Per-row center + inverse norm of merged [NR,32]; bf16 centered rows out.
__global__ void k_row_center(const float* __restrict__ M, bf16_t* __restrict__ xcb,
                             float* __restrict__ invn) {
  int wid  = uniform_wid();
  int lane = threadIdx.x & 31;
  if (wid >= NR) return;
  float v    = M[(size_t)wid * 32 + lane];
  float mean = wave_sum(v) * (1.f / 32.f);
  float xc   = v - mean;
  float ss   = wave_sum(xc * xc);
  xcb[(size_t)wid * 32 + lane] = (bf16_t)xc;
  if (lane == 0) invn[wid] = (ss > 0.f) ? rsqrtf(ss) : 0.f;
}

// HW [NR,64] f32 -> HWt [64,NR] bf16 (B operand for the aggregation WMMA)
__global__ void k_hw_t(const float* __restrict__ HW, bf16_t* __restrict__ HWt) {
  long long i = (long long)blockIdx.x * blockDim.x + threadIdx.x;
  if (i >= (long long)NR * 64) return;
  int       f = (int)(i % 64);
  long long r = i / 64;
  HWt[(size_t)f * NR + r] = (bf16_t)HW[i];
}

// Degree pass: deg[j] (pre-set to 1.0 for the self loop) += sum_i masked C[i,j].
// Each wave: one 16x16 C tile via WMMA, mask+scale in regs, column reduce.
__global__ void k_deg(const bf16_t* __restrict__ xcb, const float* __restrict__ invn,
                      float* __restrict__ deg) {
  const int wid = uniform_wid();
  if (wid >= MT * MT) return;
  int it = wid % MT, jt = wid / MT;
  int lane = threadIdx.x & 31, l16 = lane & 15, h = lane >> 4;
  v16bf a = load_frag_a(xcb + (size_t)(it * 16 + l16) * 32 + h * 16);
  v16bf b = load_frag_b(xcb + (size_t)(jt * 16 + l16) * 32 + h * 16);
  v8f c = vzero();
  c = wmma_bf16(a, b, c);
  int   j    = jt * 16 + l16;
  float inj  = invn[j];
  float part = 0.f;
#pragma unroll
  for (int e = 0; e < 8; ++e) {
    int i = it * 16 + h * 8 + e;
    float cij = c[e] * invn[i] * inj;
    int bs = (i / 1000) * 1000;                     // CHUNK = 1000
    if (fabsf(cij) > 0.8f && i != j && j >= bs + 1) part += cij;
  }
  part += __shfl_xor(part, 16, 32);                 // combine row halves
  if (h == 0) atomicAdd(&deg[j], part);
}

__global__ void k_deg_fin(const float* __restrict__ deg, float* __restrict__ dinv) {
  int i = blockIdx.x * blockDim.x + threadIdx.x;
  if (i >= NR) return;
  float d = deg[i];
  dinv[i] = (d > 0.f) ? rsqrtf(fmaxf(d, 1e-12f)) : 0.f;
}

// Fused GCN aggregation: gacc[j,:64] += sum_i P[i,j] * HW[i,:64] where
// P = masked(C)*dinv_i.  C tile via WMMA#1 (bf16), mask/scale in registers,
// LDS round-trip re-shapes P^T into an A-fragment for WMMA#2 against HWt.
__global__ void k_agg(const bf16_t* __restrict__ xcb, const float* __restrict__ invn,
                      const float* __restrict__ dinv, const bf16_t* __restrict__ HWt,
                      float* __restrict__ gacc) {
  __shared__ __align__(16) bf16_t lds[8][512];      // per-wave 16(j) x 32(i) bf16
  const int NCH = 8;                                // i-range chunks (parallelism)
  const int wid = uniform_wid();
  if (wid >= MT * NCH) return;
  int jt = wid % MT;
  int ch = wid / MT;
  int lane = threadIdx.x & 31, l16 = lane & 15, h = lane >> 4;
  bf16_t* P = &lds[threadIdx.x >> 5][0];

  v8f acc[4];
#pragma unroll
  for (int t = 0; t < 4; ++t) acc[t] = vzero();

  v16bf bj  = load_frag_b(xcb + (size_t)(jt * 16 + l16) * 32 + h * 16);
  int   j   = jt * 16 + l16;
  float inj = invn[j];

  for (int itp = ch * 32; itp < ch * 32 + 32; ++itp) {   // pairs of i-tiles
#pragma unroll
    for (int half = 0; half < 2; ++half) {
      int it = itp * 2 + half;
      v16bf ai = load_frag_a(xcb + (size_t)(it * 16 + l16) * 32 + h * 16);
      v8f c = vzero();
      c = wmma_bf16(ai, bj, c);
      // P^T[j=l16][k = half*16 + 8h + e], packed as 4x 32-bit LDS stores
      unsigned* P32 = (unsigned*)(P + l16 * 32 + half * 16 + h * 8);
#pragma unroll
      for (int ee = 0; ee < 4; ++ee) {
        union { bf16_t b2[2]; unsigned u; } pk;
#pragma unroll
        for (int s = 0; s < 2; ++s) {
          int e = ee * 2 + s;
          int i = it * 16 + h * 8 + e;
          float cij = c[e] * invn[i] * inj;
          int bs = (i / 1000) * 1000;
          float p = (fabsf(cij) > 0.8f && i != j && j >= bs + 1) ? cij * dinv[i]
                                                                 : 0.f;
          pk.b2[s] = (bf16_t)p;
        }
        P32[ee] = pk.u;
      }
    }
    // same-wave LDS RAW is in-order on CDNA5; no barrier needed
    v16bf pa = load_frag_a(P + l16 * 32 + h * 16);
#pragma unroll
    for (int t = 0; t < 4; ++t) {
      v16bf bh = load_frag_b(HWt + (size_t)(t * 16 + l16) * NR +
                             (size_t)itp * 32 + h * 16);
      acc[t] = wmma_bf16(pa, bh, acc[t]);
    }
  }
#pragma unroll
  for (int t = 0; t < 4; ++t) {
    int f = t * 16 + l16;
#pragma unroll
    for (int e = 0; e < 8; ++e) {
      int jj = jt * 16 + h * 8 + e;
      atomicAdd(&gacc[(size_t)jj * 64 + f], acc[t][e]);
    }
  }
}

// gout = dinv_j * (gacc + dinv_j * HW[j]) + gcn_bias   (adds the self loop)
__global__ void k_gcn_epi(const float* __restrict__ gacc, const float* __restrict__ dinv,
                          const float* __restrict__ HW, const float* __restrict__ gb,
                          float* __restrict__ gout) {
  long long i = (long long)blockIdx.x * blockDim.x + threadIdx.x;
  if (i >= (long long)NR * 64) return;
  int       f  = (int)(i % 64);
  long long jj = i / 64;
  float dj = dinv[jj];
  gout[i] = dj * (gacc[i] + dj * HW[i]) + gb[f];
}

// bn -> leaky_relu(0.1) -> + residual -> layernorm(64) -> final dot (64->1)
__global__ void k_post(const float* __restrict__ gout, const float* __restrict__ aff,
                       const float* __restrict__ resid, const float* __restrict__ fw,
                       const float* __restrict__ fb, float* __restrict__ out) {
  int row  = uniform_wid();
  int lane = threadIdx.x & 31;
  if (row >= NR) return;
  int f0 = lane, f1 = lane + 32;
  size_t base = (size_t)row * 64;
  float v0 = aff[f0] * gout[base + f0] + aff[256 + f0];
  float v1 = aff[f1] * gout[base + f1] + aff[256 + f1];
  v0 = (v0 > 0.f) ? v0 : 0.1f * v0;
  v1 = (v1 > 0.f) ? v1 : 0.1f * v1;
  v0 += resid[base + f0];
  v1 += resid[base + f1];
  float mean = wave_sum(v0 + v1) * (1.f / 64.f);
  float d0 = v0 - mean, d1 = v1 - mean;
  float var = wave_sum(d0 * d0 + d1 * d1) * (1.f / 64.f);
  float inv = rsqrtf(var + 1e-5f);
  float dot = wave_sum(d0 * inv * fw[f0] + d1 * inv * fw[f1]);
  if (lane == 0) out[row] = dot + fb[0];
}

// ---------------------------------------------------------------------------
// Host orchestration
// ---------------------------------------------------------------------------
extern "C" void kernel_launch(void* const* d_in, const int* in_sizes, int n_in,
                              void* d_out, int out_size, void* d_ws, size_t ws_size,
                              hipStream_t stream) {
  (void)in_sizes; (void)n_in; (void)out_size; (void)ws_size;

  char*  wsb = (char*)d_ws;
  size_t off = 0;
  auto alloc = [&](size_t bytes) -> char* {
    char* p = wsb + off;
    off += (bytes + 255) & ~(size_t)255;
    return p;
  };

  bf16_t* XB      = (bf16_t*)alloc((size_t)NR * 2048 * 2);  // bf16 GEMM input
  bf16_t* WB      = (bf16_t*)alloc((size_t)256 * 2048 * 2); // bf16 weights
  float*  T1      = (float*)alloc((size_t)NR * 256 * 4);
  float*  T2      = (float*)alloc((size_t)NR * 256 * 4);
  float*  T3      = (float*)alloc((size_t)NR * 256 * 4);
  float*  ACTF0   = (float*)alloc((size_t)NR * 256 * 4);
  float*  ACTF1   = (float*)alloc((size_t)NR * 256 * 4);
  bf16_t* ACTB0   = (bf16_t*)alloc((size_t)NR * 256 * 2);
  bf16_t* ACTB1   = (bf16_t*)alloc((size_t)NR * 256 * 2);
  bf16_t* HB      = (bf16_t*)alloc((size_t)NR * 256 * 2);
  float*  MERGED  = (float*)alloc((size_t)NR * 32 * 4);
  bf16_t* MERGEDB = (bf16_t*)alloc((size_t)NR * 32 * 2);
  bf16_t* XCB     = (bf16_t*)alloc((size_t)NR * 32 * 2);
  float*  INVN    = (float*)alloc((size_t)NR * 4);
  float*  DEG     = (float*)alloc((size_t)NR * 4);
  float*  DINV    = (float*)alloc((size_t)NR * 4);
  float*  HW      = (float*)alloc((size_t)NR * 64 * 4);
  bf16_t* HWT     = (bf16_t*)alloc((size_t)64 * NR * 2);
  float*  RESID   = (float*)alloc((size_t)NR * 64 * 4);
  float*  GACC    = (float*)alloc((size_t)NR * 64 * 4);
  float*  GOUT    = (float*)alloc((size_t)NR * 64 * 4);
  float*  SUMS    = (float*)alloc(512 * 4);
  float*  AFF1    = (float*)alloc(512 * 4);
  float*  AFF2    = (float*)alloc(512 * 4);
  float*  AFF3    = (float*)alloc(512 * 4);

  auto pin = [&](int i) { return (const float*)d_in[i]; };

  auto cvt = [&](const float* src, int srcLd, int K, int rows, bf16_t* dst, int Kp) {
    long long tot = (long long)rows * Kp;
    k_cvt_pad<<<(int)((tot + 255) / 256), 256, 0, stream>>>(src, srcLd, K, dst, Kp, tot);
  };
  // Fout is always a multiple of 64 (async-staged) except lin_out's 16.
  auto gemm = [&](const bf16_t* A, const bf16_t* W, const float* bias, float* D,
                  int Kp, int Fout, int ldD, int colOff) {
    if (Fout % 64 == 0) {
      int blocks = (MT / 8) * (Fout / 64);
      k_gemm4s<<<blocks, 256, 0, stream>>>(A, W, bias, D, Kp, ldD, colOff);
    } else {  // Fout == 16
      int waves = MT;
      k_gemm<1><<<(waves * 32 + 255) / 256, 256, 0, stream>>>(A, W, bias, D, Kp,
                                                              ldD, colOff, waves);
    }
  };
  auto bn = [&](const float* X, int F, const float* g, const float* b, float* aff) {
    k_fill<<<2, 256, 0, stream>>>(SUMS, 512, 0.f);
    k_bn_stats<<<NR / 64, F, 0, stream>>>(X, F, SUMS, SUMS + 256);
    k_bn_fin<<<1, 256, 0, stream>>>(SUMS, SUMS + 256, g, b, F, aff);
  };
  auto resblock = [&](int pb, bool sc, int fin, int fout, const float* inF,
                      const bf16_t* inB, float* outF, bf16_t* outB) {
    long long tot = (long long)NR * fout;
    int eg = (int)((tot + 255) / 256);
    cvt(pin(pb + 0), fin, fin, fout, WB, fin);
    gemm(inB, WB, pin(pb + 1), T1, fin, fout, fout, 0);
    bn(T1, fout, pin(pb + 2), pin(pb + 3), AFF1);
    k_bn_apply<<<eg, 256, 0, stream>>>(T1, AFF1, fout, tot, 1, nullptr, HB);
    cvt(pin(pb + 4), fout, fout, fout, WB, fout);
    gemm(HB, WB, pin(pb + 5), T2, fout, fout, fout, 0);
    bn(T2, fout, pin(pb + 6), pin(pb + 7), AFF2);
    if (sc) {
      cvt(pin(pb + 8), fin, fin, fout, WB, fin);
      gemm(inB, WB, pin(pb + 9), T3, fin, fout, fout, 0);
      bn(T3, fout, pin(pb + 10), pin(pb + 11), AFF3);
      k_res_tail<<<eg, 256, 0, stream>>>(T2, AFF2, T3, AFF3, fout, tot, outF, outB);
    } else {
      k_res_tail<<<eg, 256, 0, stream>>>(T2, AFF2, inF, nullptr, fout, tot, outF, outB);
    }
  };
  auto processor = [&](int base, int K, int colOff) {
    int Kp = (K + 31) & ~31;
    cvt((const float*)d_in[0], 4096, K, NR, XB, Kp);          // x slice -> bf16
    cvt(pin(base + 0), K, K, 256, WB, Kp);                    // lin_in
    gemm(XB, WB, pin(base + 1), T1, Kp, 256, 256, 0);
    bn(T1, 256, pin(base + 2), pin(base + 3), AFF1);
    long long tot = (long long)NR * 256;
    k_bn_apply<<<(int)((tot + 255) / 256), 256, 0, stream>>>(T1, AFF1, 256, tot, 0,
                                                             ACTF0, ACTB0);
    resblock(base + 4,  false, 256, 256, ACTF0, ACTB0, ACTF1, ACTB1);
    resblock(base + 12, true,  256, 128, ACTF1, ACTB1, ACTF0, ACTB0);
    resblock(base + 24, true,  128, 64,  ACTF0, ACTB0, ACTF1, ACTB1);
    cvt(pin(base + 36), 64, 64, 16, WB, 64);                  // lin_out -> merged
    gemm(ACTB1, WB, pin(base + 37), MERGED, 64, 16, 32, colOff);
  };

  // ---- two processors -> merged [NR, 32] ----
  processor(1, 1250, 0);    // main    -> cols 0..15
  processor(39, 2048, 16);  // combined-> cols 16..31

  // ---- graph stage ----
  cvt(MERGED, 32, 32, NR, MERGEDB, 32);
  cvt(pin(77), 32, 32, 64, WB, 32);                           // gcn_lin.w [64,32]
  gemm(MERGEDB, WB, nullptr, HW, 32, 64, 64, 0);              // HW = merged @ Wg^T
  cvt(pin(81), 32, 32, 64, WB, 32);                           // residual_fc.w
  gemm(MERGEDB, WB, pin(82), RESID, 32, 64, 64, 0);

  k_row_center<<<NR * 32 / 256, 256, 0, stream>>>(MERGED, XCB, INVN);
  {
    long long t = (long long)NR * 64;
    k_hw_t<<<(int)((t + 255) / 256), 256, 0, stream>>>(HW, HWT);
  }
  k_fill<<<(NR + 255) / 256, 256, 0, stream>>>(DEG, NR, 1.0f); // self-loop deg
  k_deg<<<(MT * MT * 32) / 256, 256, 0, stream>>>(XCB, INVN, DEG);
  k_deg_fin<<<(NR + 255) / 256, 256, 0, stream>>>(DEG, DINV);
  k_fill<<<(int)(((long long)NR * 64 + 255) / 256), 256, 0, stream>>>(
      GACC, (long long)NR * 64, 0.f);
  k_agg<<<(MT * 8 * 32) / 256, 256, 0, stream>>>(XCB, INVN, DINV, HWT, GACC);
  {
    long long t = (long long)NR * 64;
    k_gcn_epi<<<(int)((t + 255) / 256), 256, 0, stream>>>(GACC, DINV, HW, pin(78),
                                                          GOUT);
  }
  bn(GOUT, 64, pin(79), pin(80), AFF1);
  k_post<<<NR * 32 / 256, 256, 0, stream>>>(GOUT, AFF1, RESID, pin(83), pin(84),
                                            (float*)d_out);
}
